// DyDCNv2_45131516346612
// MI455X (gfx1250) — matrix operand order repaired and checked by reading, not compiled, not run on hardware
//
#include <hip/hip_runtime.h>
#include <hip/hip_bf16.h>

typedef __attribute__((ext_vector_type(16))) _Float16 v16h;
typedef __attribute__((ext_vector_type(8)))  float    v8f;

#define B_      8
#define C_      256
#define O_      256
#define H_      64
#define W_      64
#define HW_     4096
#define KPTS    9
#define CK      2304      // C_*KPTS
#define PTILE   16
#define KCSTEPS 72        // CK/32
#define KHALF   1152      // CK/2 per LDS stage
#define VROW    1160      // 1152 + 8 halves pad -> 4-bank lane stagger
#define GROUPS_ 16
#define EPS_    1e-5f

// ---------------------------------------------------------------------------
// Kernel 0: convert weight f32 -> f16, pre-swizzled into the WMMA A-fragment
// layout for v_wmma_f32_16x16x32_f16 in this code's contiguous k order (B panel reads the same):
//   lane<16 : M=lane,    halves 0-15 -> K 0-15
//   lane>=16: M=lane-16, halves 0-15 -> K 16-31
// Stored as afrag[((mt*72 + kc)*32 + lane)*16 + j]  (f16)
// ---------------------------------------------------------------------------
__global__ void prep_weight_kernel(const float* __restrict__ w,
                                   _Float16* __restrict__ afrag) {
    int i = blockIdx.x * blockDim.x + threadIdx.x;   // 0 .. 589823
    if (i >= O_ * CK) return;
    int j  = i & 15;
    int l  = (i >> 4) & 31;
    int kc = (i >> 9) % KCSTEPS;
    int mt = i / (KCSTEPS * 512);
    int M  = mt * 16 + (l & 15);
    int K  = kc * 32 + ((l < 16) ? 0 : 16) + j;
    afrag[i] = (_Float16)w[M * CK + K];
}

// ---------------------------------------------------------------------------
// Kernel 1: fused deformable-gather + WMMA GEMM + GN partial stats.
// One workgroup = one batch b and one row-aligned tile of 16 pixels.
// 8 waves x 2 M-tiles = all 256 output channels.
// ---------------------------------------------------------------------------
__global__ __launch_bounds__(256) void dcn_wmma_kernel(
    const float* __restrict__ x, const float* __restrict__ off,
    const float* __restrict__ mask, const _Float16* __restrict__ afrag,
    float* __restrict__ out, float* __restrict__ stats)
{
    __shared__ __align__(16) _Float16 vpanel[PTILE * VROW];   // 37120 B
    __shared__ int   pidx[KPTS * PTILE * 4];                  // 2304 B
    __shared__ float pwgt[KPTS * PTILE * 4];                  // 2304 B

    const int tid = threadIdx.x;
    const int b   = blockIdx.x >> 8;        // /256 pixel-tiles per batch
    const int pb  = blockIdx.x & 255;
    const int p0  = pb * PTILE;
    const int h   = p0 >> 6;                // W_=64, 16 | 64 -> same row
    const int w0  = p0 & 63;

    // ---- Phase A: bilinear parameters for 9 kernel points x 16 pixels ----
    if (tid < KPTS * PTILE) {
        int k = tid >> 4;
        int n = tid & 15;
        int wp = w0 + n;
        int hw = h * W_ + wp;
        float dy = off[((size_t)(b * 18 + 2 * k    )) * HW_ + hw];
        float dx = off[((size_t)(b * 18 + 2 * k + 1)) * HW_ + hw];
        float m  = mask[((size_t)(b * KPTS + k)) * HW_ + hw];
        float ys = (float)h  + (float)(k / 3 - 1) + dy;
        float xs = (float)wp + (float)(k % 3 - 1) + dx;
        float y0f = floorf(ys), x0f = floorf(xs);
        float wy1 = ys - y0f,  wx1 = xs - x0f;
        int y0 = (int)y0f, x0 = (int)x0f;
#pragma unroll
        for (int c4 = 0; c4 < 4; ++c4) {
            int iy = c4 >> 1, ix = c4 & 1;
            int yy = y0 + iy, xx = x0 + ix;
            bool valid = (yy >= 0) && (yy < H_) && (xx >= 0) && (xx < W_);
            int yc = min(max(yy, 0), H_ - 1);
            int xc = min(max(xx, 0), W_ - 1);
            float wgt = (iy ? wy1 : 1.0f - wy1) * (ix ? wx1 : 1.0f - wx1);
            pidx[tid * 4 + c4] = yc * W_ + xc;
            pwgt[tid * 4 + c4] = valid ? wgt * m : 0.0f;
        }
    }
    __syncthreads();

    const float* xb = x + (size_t)b * C_ * HW_;
    const int wave = tid >> 5;
    const int lane = tid & 31;
    const int nloc = lane & 15;
    const int hi   = (lane >> 4) & 1;
    const int mt0  = wave * 2;
    const int mt1  = wave * 2 + 1;

    v8f acc0 = {}; v8f acc1 = {};
    const _Float16* bbase = vpanel + nloc * VROW + hi * 16;
    const _Float16* a0 = afrag + ((size_t)(mt0 * KCSTEPS) * 32 + lane) * 16;
    const _Float16* a1 = afrag + ((size_t)(mt1 * KCSTEPS) * 32 + lane) * 16;

    // ---- two LDS stages of 1152 K-values each (keeps static LDS < 64KB) ----
    for (int s = 0; s < 2; ++s) {
        // gather stage: 1152*16 = 18432 f16 entries, 72 per thread
        for (int i = 0; i < (KHALF * PTILE) / 256; ++i) {
            int e   = i * 256 + tid;
            int n   = e / KHALF;
            int ckl = e - n * KHALF;
            int ck  = s * KHALF + ckl;
            int c   = ck / 9;
            int k   = ck - c * 9;
            int pp  = (k * 16 + n) * 4;
            const float* xc = xb + (size_t)c * HW_;
            float v = pwgt[pp + 0] * xc[pidx[pp + 0]]
                    + pwgt[pp + 1] * xc[pidx[pp + 1]]
                    + pwgt[pp + 2] * xc[pidx[pp + 2]]
                    + pwgt[pp + 3] * xc[pidx[pp + 3]];
            vpanel[n * VROW + ckl] = (_Float16)v;
        }
        __syncthreads();

        // WMMA stage: 36 K-chunks of 32
        const int kc0 = s * (KCSTEPS / 2);
        for (int kcl = 0; kcl < KCSTEPS / 2; ++kcl) {
            int kc = kc0 + kcl;
            v16h bf  = *(const v16h*)(bbase + kcl * 32);
            v16h af0 = *(const v16h*)(a0 + (size_t)kc * 512);
            v16h af1 = *(const v16h*)(a1 + (size_t)kc * 512);
            acc0 = __builtin_amdgcn_wmma_f32_16x16x32_f16(
                false, af0, false, bf, (short)0, acc0, false, false);
            acc1 = __builtin_amdgcn_wmma_f32_16x16x32_f16(
                false, af1, false, bf, (short)0, acc1, false, false);
        }
        __syncthreads();
    }

    // ---- store conv result + per-group (== per M-tile) stats ----
    float* outb = out + (size_t)b * O_ * HW_ + p0 + nloc;
    float s0 = 0.f, q0 = 0.f, s1 = 0.f, q1 = 0.f;
#pragma unroll
    for (int r = 0; r < 8; ++r) {
        int M0 = mt0 * 16 + r + hi * 8;
        int M1 = mt1 * 16 + r + hi * 8;
        float v0 = acc0[r], v1 = acc1[r];
        outb[(size_t)M0 * HW_] = v0;
        outb[(size_t)M1 * HW_] = v1;
        s0 += v0; q0 += v0 * v0;
        s1 += v1; q1 += v1 * v1;
    }
    for (int m = 16; m >= 1; m >>= 1) {
        s0 += __shfl_xor(s0, m, 32); q0 += __shfl_xor(q0, m, 32);
        s1 += __shfl_xor(s1, m, 32); q1 += __shfl_xor(q1, m, 32);
    }
    if (lane == 0) {
        atomicAdd(&stats[(b * GROUPS_ + mt0) * 2 + 0], s0);
        atomicAdd(&stats[(b * GROUPS_ + mt0) * 2 + 1], q0);
        atomicAdd(&stats[(b * GROUPS_ + mt1) * 2 + 0], s1);
        atomicAdd(&stats[(b * GROUPS_ + mt1) * 2 + 1], q1);
    }
}

// ---------------------------------------------------------------------------
// Kernel 2: GroupNorm finalize, in-place on d_out.
// ---------------------------------------------------------------------------
__global__ __launch_bounds__(256) void gn_finalize_kernel(
    float* __restrict__ out, const float* __restrict__ stats,
    const float* __restrict__ gamma, const float* __restrict__ beta)
{
    size_t i = (size_t)blockIdx.x * blockDim.x + threadIdx.x;
    int o = (int)((i >> 12) & 255);         // / HW_ % O_
    int b = (int)(i >> 20);                 // / (O_*HW_)
    int g = o >> 4;                         // 16 channels per group
    float sum = stats[(b * GROUPS_ + g) * 2 + 0];
    float ssq = stats[(b * GROUPS_ + g) * 2 + 1];
    const float invN = 1.0f / (float)(16 * HW_);   // 65536 elems per group
    float mean = sum * invN;
    float var  = ssq * invN - mean * mean;
    float sc   = rsqrtf(var + EPS_);
    float v = out[i];
    out[i] = (v - mean) * sc * gamma[o] + beta[o];
}

// ---------------------------------------------------------------------------
extern "C" void kernel_launch(void* const* d_in, const int* in_sizes, int n_in,
                              void* d_out, int out_size, void* d_ws, size_t ws_size,
                              hipStream_t stream) {
    const float* x      = (const float*)d_in[0];
    const float* off    = (const float*)d_in[1];
    const float* mask   = (const float*)d_in[2];
    const float* weight = (const float*)d_in[3];
    const float* gamma  = (const float*)d_in[4];
    const float* beta   = (const float*)d_in[5];
    float* out = (float*)d_out;

    float*    stats = (float*)d_ws;                        // 256 floats
    _Float16* afrag = (_Float16*)((char*)d_ws + 1024);     // 1.18 MB swizzled W

    hipMemsetAsync(stats, 0, B_ * GROUPS_ * 2 * sizeof(float), stream);

    prep_weight_kernel<<<(O_ * CK + 255) / 256, 256, 0, stream>>>(weight, afrag);

    dcn_wmma_kernel<<<B_ * (HW_ / PTILE), 256, 0, stream>>>(
        x, off, mask, afrag, out, stats);

    gn_finalize_kernel<<<(B_ * O_ * HW_) / 256, 256, 0, stream>>>(
        out, stats, gamma, beta);
}
